// FeedForwardNet_39762807226554
// MI455X (gfx1250) — compile-verified, matrix operand order checked by reading, not run on hardware
//
#include <hip/hip_runtime.h>

typedef _Float16 v16h __attribute__((ext_vector_type(16)));
typedef _Float16 v8h  __attribute__((ext_vector_type(8)));
typedef _Float16 v4h  __attribute__((ext_vector_type(4)));
typedef _Float16 v2h  __attribute__((ext_vector_type(2)));
typedef float    v8f  __attribute__((ext_vector_type(8)));

#define D_FEAT 512
#define H_FEAT 100
#define O_FEAT 10
#define BLOCK_ROWS 128
#define THREADS 256
#define NT1 7              // ceil(100/16) N-tiles for GEMM1
#define KT1 16             // 512/32 K-steps for GEMM1
#define KT2 4              // 128/32 K-steps for GEMM2 (H padded 100->128)
#define W1PACK_DW (KT1 * NT1 * 32 * 8)   // 28672 dwords
#define W2PACK_DW (KT2 * 32 * 8)         // 1024 dwords

// K index held in VGPR v (pair base) for 16-bit A/B fragments, per CDNA5 ISA
// 16x32 f16 layout: lanes 0-15 -> K={0..7,16..23}, lanes 16-31 -> +8.
__device__ __host__ __forceinline__ int kfrag(int v, int half) {
  return (v < 4) ? (2 * v + 8 * half) : (16 + 2 * (v - 4) + 8 * half);
}

// ---------------------------------------------------------------------------
// Prep: repack W1 (512x100 f32) and W2 (100x10 f32) into fragment-ready f16
// layouts so each lane loads one contiguous 32B B-fragment per WMMA.
// w1p dword index = ((kt*7 + nt)*32 + lane)*8 + v  -> halves (K=k0,k0+1; N=n)
// ---------------------------------------------------------------------------
__global__ void pack_weights_kernel(const float* __restrict__ W1,
                                    const float* __restrict__ W2,
                                    unsigned int* __restrict__ w1p,
                                    unsigned int* __restrict__ w2p) {
  int idx = blockIdx.x * blockDim.x + threadIdx.x;
  if (idx < W1PACK_DW) {
    int v    = idx & 7;
    int lane = (idx >> 3) & 31;
    int tile = idx >> 8;          // kt*7 + nt
    int nt   = tile % NT1;
    int kt   = tile / NT1;
    int n    = nt * 16 + (lane & 15);
    int half = lane >> 4;
    int k0   = 32 * kt + kfrag(v, half);
    float e0 = (n < H_FEAT) ? W1[k0 * H_FEAT + n] : 0.f;
    float e1 = (n < H_FEAT) ? W1[(k0 + 1) * H_FEAT + n] : 0.f;
    union { v2h h; unsigned int u; } p;
    p.h[0] = (_Float16)e0;
    p.h[1] = (_Float16)e1;
    w1p[idx] = p.u;
  } else {
    int i2 = idx - W1PACK_DW;
    if (i2 < W2PACK_DW) {
      int v    = i2 & 7;
      int lane = (i2 >> 3) & 31;
      int kt   = i2 >> 8;         // 0..3
      int n    = lane & 15;
      int half = lane >> 4;
      int k0   = 32 * kt + kfrag(v, half);
      float e0 = (k0     < H_FEAT && n < O_FEAT) ? W2[k0 * O_FEAT + n]       : 0.f;
      float e1 = (k0 + 1 < H_FEAT && n < O_FEAT) ? W2[(k0 + 1) * O_FEAT + n] : 0.f;
      union { v2h h; unsigned int u; } p;
      p.h[0] = (_Float16)e0;
      p.h[1] = (_Float16)e1;
      w2p[i2] = p.u;
    }
  }
}

// ---------------------------------------------------------------------------
// Fused LayerNorm -> Linear(512,100) -> SiLU -> Linear(100,10).
// 256 threads = 8 waves; 128 rows per block; one 16-row WMMA tile per wave.
// 128 KB LDS -> 2 blocks/WGP = 4 waves/SIMD (LDS-capped). Tell the register
// allocator that via waves_per_eu so it stops recycling B-fragment buffers.
// ---------------------------------------------------------------------------
__global__ __launch_bounds__(THREADS)
__attribute__((amdgpu_waves_per_eu(4, 4)))
void ffn_fused_kernel(const float* __restrict__ x,
                      const float* __restrict__ gamma,
                      const float* __restrict__ beta,
                      const float* __restrict__ b1,
                      const float* __restrict__ b2,
                      const unsigned int* __restrict__ w1p,
                      const unsigned int* __restrict__ w2p,
                      float* __restrict__ out) {
  extern __shared__ _Float16 smem[];   // BLOCK_ROWS * 512 halves = 128 KB
  const int t          = threadIdx.x;
  const int block_row0 = blockIdx.x * BLOCK_ROWS;

  // ---------------- Phase 1: LayerNorm, xn (f16) -> LDS ----------------
  {
    int rl  = t >> 1;                 // local row 0..127
    int sub = t & 1;                  // which half of the row
    const float* px = x + (size_t)(block_row0 + rl) * D_FEAT + sub * 256;
    float s1 = 0.f, s2 = 0.f;
    #pragma unroll 8
    for (int i = 0; i < 64; ++i) {
      float4 v = ((const float4*)px)[i];
      s1 += v.x + v.y + v.z + v.w;
      s2 += v.x * v.x + v.y * v.y + v.z * v.z + v.w * v.w;
    }
    s1 += __shfl_xor(s1, 1, 32);      // partner lane handles other half-row
    s2 += __shfl_xor(s2, 1, 32);
    float mean = s1 * (1.f / D_FEAT);
    float var  = s2 * (1.f / D_FEAT) - mean * mean;
    float rstd = __frsqrt_rn(var + 1e-5f);
    const float4* pg = (const float4*)(gamma + sub * 256);
    const float4* pb = (const float4*)(beta  + sub * 256);
    _Float16* dst = smem + rl * D_FEAT + sub * 256;
    #pragma unroll 8
    for (int i = 0; i < 64; ++i) {
      float4 v = ((const float4*)px)[i];   // re-read: guaranteed L2 hit
      float4 g = pg[i];
      float4 bb = pb[i];
      v4h o;
      o[0] = (_Float16)((v.x - mean) * rstd * g.x + bb.x);
      o[1] = (_Float16)((v.y - mean) * rstd * g.y + bb.y);
      o[2] = (_Float16)((v.z - mean) * rstd * g.z + bb.z);
      o[3] = (_Float16)((v.w - mean) * rstd * g.w + bb.w);
      *(v4h*)(dst + i * 4) = o;
    }
  }
  __syncthreads();

  // ---------------- Phase 2: GEMM1 via v_wmma_f32_16x16x32_f16 ----------------
  const int w    = t >> 5;        // wave id: owns rows 16w..16w+15
  const int lane = t & 31;
  const int half = lane >> 4;
  const int n16  = lane & 15;     // N column within tile / A row within tile
  _Float16* xrow = smem + (w * 16 + n16) * D_FEAT;

  float biasv[NT1];
  #pragma unroll
  for (int nt = 0; nt < NT1; ++nt) {
    int n = nt * 16 + n16;
    biasv[nt] = (n < H_FEAT) ? b1[n] : 0.f;
  }

  v8f acc[NT1];
  #pragma unroll
  for (int nt = 0; nt < NT1; ++nt) acc[nt] = (v8f){};

  const v16h* w1f = (const v16h*)w1p;
  // Fully unrolled K loop: with the relaxed register budget the scheduler can
  // hoist whole 14-load clauses across WMMA groups (cross-K software pipeline).
  #pragma unroll
  for (int kt = 0; kt < KT1; ++kt) {
    // A fragment: two contiguous 16B LDS chunks per lane (documented layout)
    v8h alo = *(const v8h*)(xrow + kt * 32 + 8 * half);
    v8h ahi = *(const v8h*)(xrow + kt * 32 + 16 + 8 * half);
    v16h a;
    #pragma unroll
    for (int i = 0; i < 8; ++i) { a[i] = alo[i]; a[8 + i] = ahi[i]; }

    // Batch-load ALL 7 B fragments (one clause of 14 global_load_b128).
    v16h bfrag[NT1];
    #pragma unroll
    for (int nt = 0; nt < NT1; ++nt)
      bfrag[nt] = w1f[(kt * NT1 + nt) * 32 + lane];   // one 32B load / lane

    #pragma unroll
    for (int nt = 0; nt < NT1; ++nt)
      acc[nt] = __builtin_amdgcn_wmma_f32_16x16x32_f16(
          false, a, false, bfrag[nt], (short)0, acc[nt], false, false);
  }

  // ---------------- Epilogue 1: bias + SiLU, stage h into own LDS region ----
  // Reuse this wave's (now consumed) xn rows: 16 rows x 128 halves needed.
  _Float16* hbase = smem + (w * 16) * D_FEAT;
  #pragma unroll
  for (int nt = 0; nt < NT1; ++nt) {
    #pragma unroll
    for (int r = 0; r < 8; ++r) {
      float vv = acc[nt][r] + biasv[nt];
      float s  = vv / (1.f + __expf(-vv));         // SiLU; padded cols stay 0
      hbase[(r + 8 * half) * 128 + nt * 16 + n16] = (_Float16)s;
    }
  }
  #pragma unroll
  for (int r = 0; r < 8; ++r)                      // zero K-pad cols 112..127
    hbase[(r + 8 * half) * 128 + 112 + n16] = (_Float16)0.f;

  // DS ops are in-order within a wave; fence the compiler and LDS counter.
  asm volatile("s_wait_dscnt 0" ::: "memory");

  // ---------------- Phase 3: GEMM2 (16x128x16, 4 WMMAs) ----------------
  const v16h* w2f = (const v16h*)w2p;
  _Float16* hrow = hbase + n16 * 128;

  // Batch all 4 A and 4 B fragments, then issue the 4 WMMAs back-to-back.
  v16h a2[KT2];
  v16h b2f[KT2];
  #pragma unroll
  for (int kt = 0; kt < KT2; ++kt) {
    v8h alo = *(const v8h*)(hrow + kt * 32 + 8 * half);
    v8h ahi = *(const v8h*)(hrow + kt * 32 + 16 + 8 * half);
    #pragma unroll
    for (int i = 0; i < 8; ++i) { a2[kt][i] = alo[i]; a2[kt][8 + i] = ahi[i]; }
    b2f[kt] = w2f[kt * 32 + lane];
  }
  v8f acc2 = (v8f){};
  #pragma unroll
  for (int kt = 0; kt < KT2; ++kt)
    acc2 = __builtin_amdgcn_wmma_f32_16x16x32_f16(
        false, a2[kt], false, b2f[kt], (short)0, acc2, false, false);

  // ---------------- Epilogue 2: bias + store 16x10 tile ----------------
  if (n16 < O_FEAT) {
    float bb = b2[n16];
    #pragma unroll
    for (int r = 0; r < 8; ++r) {
      int row = block_row0 + w * 16 + r + 8 * half;
      out[(size_t)row * O_FEAT + n16] = acc2[r] + bb;
    }
  }
}

// ---------------------------------------------------------------------------
extern "C" void kernel_launch(void* const* d_in, const int* in_sizes, int n_in,
                              void* d_out, int out_size, void* d_ws, size_t ws_size,
                              hipStream_t stream) {
  const float* x     = (const float*)d_in[0];
  const float* gamma = (const float*)d_in[1];
  const float* beta  = (const float*)d_in[2];
  const float* W1    = (const float*)d_in[3];
  const float* b1    = (const float*)d_in[4];
  const float* W2    = (const float*)d_in[5];
  const float* b2    = (const float*)d_in[6];
  float* out         = (float*)d_out;

  unsigned int* w1p = (unsigned int*)d_ws;
  unsigned int* w2p = w1p + W1PACK_DW;

  int packN = W1PACK_DW + W2PACK_DW;
  pack_weights_kernel<<<(packN + 255) / 256, 256, 0, stream>>>(W1, W2, w1p, w2p);

  int rows   = in_sizes[0] / D_FEAT;         // 262144
  int blocks = rows / BLOCK_ROWS;            // 2048
  size_t smem_bytes = (size_t)BLOCK_ROWS * D_FEAT * sizeof(_Float16);  // 128 KB
  ffn_fused_kernel<<<blocks, THREADS, smem_bytes, stream>>>(
      x, gamma, beta, b1, b2, w1p, w2p, out);
}
